// GAT_layer_18013092839763
// MI455X (gfx1250) — matrix operand order repaired and checked by reading, not compiled, not run on hardware
//
#include <hip/hip_runtime.h>
#include <float.h>

typedef __attribute__((ext_vector_type(16))) _Float16 v16h;
typedef __attribute__((ext_vector_type(8)))  _Float16 v8h;
typedef __attribute__((ext_vector_type(8)))  float    v8f;

#define N_NODES 100000
#define N_EDGES 1600000
#define DIM     128
#define HEADS   4
#define LDS_LD  (DIM + 8)   // half-words; row stride 272B keeps 16B alignment + kills bank conflicts

__device__ __forceinline__ float leakyf(float x, float s) { return x >= 0.f ? x : s * x; }

// ---- WMMA fragment loaders (CDNA5 16x16x32 f16, wave32; ISA 7.12.2 layouts) ----

// A: 16x32 (MxK). lane: row = lane&15, k0 = (lane>>4)*8; halves 0..7 -> K=k0+j, 8..15 -> K=k0+16+j
__device__ __forceinline__ v16h load_frag_A(const _Float16* base, int ld, int kk, int lane) {
    const int r  = lane & 15;
    const int k0 = (lane >> 4) * 8;
    const _Float16* p = base + r * ld + kk + k0;
    v8h lo = *(const v8h*)(p);        // 16B aligned -> b128
    v8h hi = *(const v8h*)(p + 16);
    v16h a;
#pragma unroll
    for (int j = 0; j < 8; ++j) { a[j] = lo[j]; a[8 + j] = hi[j]; }
    return a;
}

// B: 32x16 (KxN) taken from W stored row-major [out_col][k] (i.e. B[k][c] = W[c][k]).
// lane: col = lane&15, kb = (lane>>4)*8; halves 0..7 -> K=kb+j, 8..15 -> K=kb+16+j (matches A).
__device__ __forceinline__ v16h load_frag_B(const _Float16* W, int colBase, int kk, int lane) {
    const int c  = lane & 15;
    const int kb = (lane >> 4) * 8;
    const _Float16* p = W + (size_t)(colBase + c) * DIM + kk + kb;
    v8h lo = *(const v8h*)(p);
    v8h hi = *(const v8h*)(p + 16);
    v16h b;
#pragma unroll
    for (int j = 0; j < 8; ++j) { b[j] = lo[j]; b[8 + j] = hi[j]; }
    return b;
}

// ---------------- kernel 0: fp32 -> f16 weight conversion ----------------
__global__ void k_convert_weights(const float* __restrict__ W1, const float* __restrict__ Wg,
                                  const float* __restrict__ W2,
                                  _Float16* __restrict__ W1h, _Float16* __restrict__ Wgh,
                                  _Float16* __restrict__ W2h) {
    int i = blockIdx.x * blockDim.x + threadIdx.x;
    if (i < DIM * DIM) {
        W1h[i] = (_Float16)W1[i];
        Wgh[i] = (_Float16)Wg[i];
        W2h[i] = (_Float16)W2[i];
    }
}

// ---------------- kernel 1: init accumulators ----------------
__global__ void k_init(float* __restrict__ a_src, float* __restrict__ a_dst,
                       float* __restrict__ denom, float* __restrict__ amax,
                       float* __restrict__ out_accum) {
    int i = blockIdx.x * blockDim.x + threadIdx.x;
    if (i < N_NODES * DIM) out_accum[i] = 0.f;
    if (i < N_NODES * HEADS) {
        a_src[i] = 0.f; a_dst[i] = 0.f; denom[i] = 0.f; amax[i] = -FLT_MAX;
    }
}

// ------- kernel 2: fused  y = leaky(x)@W1^T + b1 ;  h = y@Wg^T ;  a_src/a_dst dots -------
__global__ __launch_bounds__(256) void k_lin1_gat_pre(
        const float* __restrict__ x, const float* __restrict__ b1,
        const _Float16* __restrict__ W1h, const _Float16* __restrict__ Wgh,
        const float* __restrict__ att_src, const float* __restrict__ att_dst,
        float* __restrict__ hbuf, float* __restrict__ a_src, float* __restrict__ a_dst) {
    __shared__ __align__(16) _Float16 tile[16][LDS_LD];
    const int nodeBase = blockIdx.x * 16;
    const int tid  = threadIdx.x;
    const int wave = tid >> 5;      // column tile 0..7
    const int lane = tid & 31;
    const int ct   = wave;

    __builtin_prefetch(W1h + (size_t)ct * 16 * DIM, 0, 3);
    __builtin_prefetch(Wgh + (size_t)ct * 16 * DIM, 0, 3);

    // stage leaky(x) tile (16x128) into LDS as f16
    for (int i = tid; i < 16 * DIM; i += 256) {
        int r = i >> 7, c = i & (DIM - 1);
        tile[r][c] = (_Float16)leakyf(x[(size_t)(nodeBase + r) * DIM + c], 0.01f);
    }
    __syncthreads();

    // GEMM1: y tile (16x16)
    v8f acc = {};
#pragma unroll
    for (int kk = 0; kk < DIM; kk += 32) {
        v16h a = load_frag_A(&tile[0][0], LDS_LD, kk, lane);
        v16h b = load_frag_B(W1h, ct * 16, kk, lane);
        acc = __builtin_amdgcn_wmma_f32_16x16x32_f16(false, a, false, b, (short)0, acc, false, false);
    }
    __syncthreads();   // everyone done reading the x tile

    // write y + b1 back into the LDS tile as f16 (each wave owns a disjoint 16-col slab)
    {
        const int c = lane & 15, hi = lane >> 4;
        const float bv = b1[ct * 16 + c];
#pragma unroll
        for (int v = 0; v < 8; ++v)
            tile[v + 8 * hi][ct * 16 + c] = (_Float16)(acc[v] + bv);
    }
    __syncthreads();

    // GEMM2: h tile (16x16)
    v8f hacc = {};
#pragma unroll
    for (int kk = 0; kk < DIM; kk += 32) {
        v16h a = load_frag_A(&tile[0][0], LDS_LD, kk, lane);
        v16h b = load_frag_B(Wgh, ct * 16, kk, lane);
        hacc = __builtin_amdgcn_wmma_f32_16x16x32_f16(false, a, false, b, (short)0, hacc, false, false);
    }

    // store h; butterfly-reduce attention dot products per row, atomically add per (node, head)
    {
        const int c   = lane & 15, hi = lane >> 4;
        const int col = ct * 16 + c;
        const int head = col >> 5;                 // uniform per wave (16-col tile within a 32-col head)
        const float attS = att_src[col];           // att_src flat [H*C] == [col]
        const float attD = att_dst[col];
#pragma unroll
        for (int v = 0; v < 8; ++v) {
            const int row = nodeBase + v + 8 * hi;
            const float hv = hacc[v];
            hbuf[(size_t)row * DIM + col] = hv;
            float as = hv * attS, ad = hv * attD;
#pragma unroll
            for (int off = 1; off < 16; off <<= 1) {
                as += __shfl_xor(as, off, 32);
                ad += __shfl_xor(ad, off, 32);
            }
            if (c == 0) {
                atomicAdd(&a_src[row * HEADS + head], as);
                atomicAdd(&a_dst[row * HEADS + head], ad);
            }
        }
    }
}

// ---------------- kernel 3: edge alpha + segment max (float max via int/uint trick) ----------------
__global__ void k_edge_alpha(const int* __restrict__ ei, const float* __restrict__ a_src,
                             const float* __restrict__ a_dst,
                             float* __restrict__ alpha, int* __restrict__ amax_i) {
    int i = blockIdx.x * blockDim.x + threadIdx.x;
    if (i >= N_EDGES * HEADS) return;
    const int e = i >> 2, hh = i & 3;
    const int s = ei[e];
    const int d = ei[N_EDGES + e];
    const float al = leakyf(a_src[s * HEADS + hh] + a_dst[d * HEADS + hh], 0.2f);
    alpha[i] = al;
    int* addr = &amax_i[d * HEADS + hh];
    if (al >= 0.f) atomicMax(addr, __float_as_int(al));
    else           atomicMin((unsigned int*)addr, __float_as_uint(al));
}

// ---------------- kernel 4: ex = exp(alpha - amax[dst]); denom += ex ----------------
__global__ void k_edge_exp(const int* __restrict__ ei, const float* __restrict__ amax,
                           float* __restrict__ alpha /* in: alpha, out: ex */,
                           float* __restrict__ denom) {
    int i = blockIdx.x * blockDim.x + threadIdx.x;
    if (i >= N_EDGES * HEADS) return;
    const int e = i >> 2, hh = i & 3;
    const int d = ei[N_EDGES + e];
    const float ex = __expf(alpha[i] - amax[d * HEADS + hh]);
    alpha[i] = ex;
    atomicAdd(&denom[d * HEADS + hh], ex);
}

// ---------------- kernel 5: message scatter: out[dst] += h[src] * coeff ----------------
__global__ __launch_bounds__(256) void k_edge_scatter(
        const int* __restrict__ ei, const float* __restrict__ ex, const float* __restrict__ denom,
        const float* __restrict__ hbuf, float* __restrict__ out_accum) {
    const int gid = blockIdx.x * blockDim.x + threadIdx.x;
    const int e = gid >> 5;                 // one wave per edge
    if (e >= N_EDGES) return;
    const int lane = gid & 31;
    const int s = ei[e];
    const int d = ei[N_EDGES + e];
    const int col  = lane * 4;              // 4 contiguous channels per lane
    const int head = col >> 5;
    const float coeff = ex[e * HEADS + head] / (denom[d * HEADS + head] + 1e-16f);
    const float4 hv = *(const float4*)(hbuf + (size_t)s * DIM + col);
    float* op = out_accum + (size_t)d * DIM + col;
    atomicAdd(op + 0, hv.x * coeff);
    atomicAdd(op + 1, hv.y * coeff);
    atomicAdd(op + 2, hv.z * coeff);
    atomicAdd(op + 3, hv.w * coeff);
}

// ---------------- kernel 6: out = leaky(msg + bias_g) @ W2^T + b2 ----------------
__global__ __launch_bounds__(256) void k_final(
        const float* __restrict__ out_accum, const float* __restrict__ bias_g,
        const _Float16* __restrict__ W2h, const float* __restrict__ b2,
        float* __restrict__ out) {
    __shared__ __align__(16) _Float16 tile[16][LDS_LD];
    const int nodeBase = blockIdx.x * 16;
    const int tid  = threadIdx.x;
    const int wave = tid >> 5;
    const int lane = tid & 31;
    const int ct   = wave;

    __builtin_prefetch(W2h + (size_t)ct * 16 * DIM, 0, 3);

    for (int i = tid; i < 16 * DIM; i += 256) {
        int r = i >> 7, c = i & (DIM - 1);
        tile[r][c] = (_Float16)leakyf(out_accum[(size_t)(nodeBase + r) * DIM + c] + bias_g[c], 0.01f);
    }
    __syncthreads();

    v8f acc = {};
#pragma unroll
    for (int kk = 0; kk < DIM; kk += 32) {
        v16h a = load_frag_A(&tile[0][0], LDS_LD, kk, lane);
        v16h b = load_frag_B(W2h, ct * 16, kk, lane);
        acc = __builtin_amdgcn_wmma_f32_16x16x32_f16(false, a, false, b, (short)0, acc, false, false);
    }

    {
        const int c = lane & 15, hi = lane >> 4;
        const int col = ct * 16 + c;
        const float bv = b2[col];
#pragma unroll
        for (int v = 0; v < 8; ++v) {
            const int row = nodeBase + v + 8 * hi;
            out[(size_t)row * DIM + col] = acc[v] + bv;
        }
    }
}

// ---------------- launch ----------------
extern "C" void kernel_launch(void* const* d_in, const int* in_sizes, int n_in,
                              void* d_out, int out_size, void* d_ws, size_t ws_size,
                              hipStream_t stream) {
    (void)in_sizes; (void)n_in; (void)out_size; (void)ws_size;
    const float* x        = (const float*)d_in[0];
    const int*   ei       = (const int*)  d_in[1];
    // d_in[2] = edge_type (unused by forward)
    const float* W1       = (const float*)d_in[3];
    const float* b1       = (const float*)d_in[4];
    const float* Wg       = (const float*)d_in[5];
    const float* att_src  = (const float*)d_in[6];
    const float* att_dst  = (const float*)d_in[7];
    const float* bias_g   = (const float*)d_in[8];
    const float* W2       = (const float*)d_in[9];
    const float* b2       = (const float*)d_in[10];
    float* out = (float*)d_out;

    // workspace carve-up (256B aligned)
    size_t off = 0;
    auto carve = [&](size_t bytes) {
        void* p = (char*)d_ws + off;
        off += (bytes + 255) & ~(size_t)255;
        return p;
    };
    _Float16* W1h      = (_Float16*)carve(DIM * DIM * sizeof(_Float16));
    _Float16* Wgh      = (_Float16*)carve(DIM * DIM * sizeof(_Float16));
    _Float16* W2h      = (_Float16*)carve(DIM * DIM * sizeof(_Float16));
    float*    hbuf     = (float*)carve((size_t)N_NODES * DIM * sizeof(float));
    float*    out_acc  = (float*)carve((size_t)N_NODES * DIM * sizeof(float));
    float*    a_src    = (float*)carve((size_t)N_NODES * HEADS * sizeof(float));
    float*    a_dst    = (float*)carve((size_t)N_NODES * HEADS * sizeof(float));
    float*    amax     = (float*)carve((size_t)N_NODES * HEADS * sizeof(float));
    float*    denom    = (float*)carve((size_t)N_NODES * HEADS * sizeof(float));
    float*    alpha    = (float*)carve((size_t)N_EDGES * HEADS * sizeof(float));

    k_convert_weights<<<(DIM * DIM + 255) / 256, 256, 0, stream>>>(W1, Wg, W2, W1h, Wgh, W2h);
    k_init<<<(N_NODES * DIM + 255) / 256, 256, 0, stream>>>(a_src, a_dst, denom, amax, out_acc);
    k_lin1_gat_pre<<<N_NODES / 16, 256, 0, stream>>>(x, b1, W1h, Wgh, att_src, att_dst,
                                                     hbuf, a_src, a_dst);
    k_edge_alpha<<<(N_EDGES * HEADS + 255) / 256, 256, 0, stream>>>(ei, a_src, a_dst, alpha, (int*)amax);
    k_edge_exp<<<(N_EDGES * HEADS + 255) / 256, 256, 0, stream>>>(ei, amax, alpha, denom);
    k_edge_scatter<<<(N_EDGES * 32 + 255) / 256, 256, 0, stream>>>(ei, alpha, denom, hbuf, out_acc);
    k_final<<<N_NODES / 16, 256, 0, stream>>>(out_acc, bias_g, W2h, b2, out);
}
